// GlobalLocalCrossAttention_85547158602262
// MI455X (gfx1250) — compile-verified, matrix operand order checked by reading, not run on hardware
//
#include <hip/hip_runtime.h>

typedef __attribute__((ext_vector_type(16))) _Float16 v16h;
typedef __attribute__((ext_vector_type(8)))  float    v8f;

#define B_   16
#define S_   2048
#define D_   768
#define H_   12
#define DK_  64
#define NL_  307            // NUM_LOCAL
#define LT_  20             // ceil(307/16)
#define MQ_  (B_ * NL_)     // 4912 = 16*307, multiple of 16

union U8h { uint4 u; _Float16 h[8]; };

// ---- WMMA 16x32 f16 tile loaders -------------------------------------------
// A layout (16-bit A 16x32): lanes 0-15 -> M=lane, elems 0..7 = K 0..7,
// elems 8..15 = K 16..23; lanes 16-31 -> same M, K ranges +8.
// B layout is symmetric with N in the lane role, so one loader serves both.

// Fast path: K contiguous in memory -> two 16B loads per lane
// (global_load_b128 or ds_load_b128 depending on address space).
__device__ inline v16h load_tile_k1(const _Float16* base, int laneStride, int lane) {
    int r    = lane & 15;
    int ksel = (lane >> 4) << 3;     // 0 or 8
    const _Float16* p = base + (long)r * laneStride + ksel;
    U8h a, b;
    a.u = *(const uint4*)(p);        // K ksel+0..7
    b.u = *(const uint4*)(p + 16);   // K ksel+16..23
    v16h out;
#pragma unroll
    for (int i = 0; i < 8; ++i) { out[i] = a.h[i]; out[i + 8] = b.h[i]; }
    return out;
}

// Generic path: arbitrary strides + lane-dim bound (zero fill).
__device__ inline v16h load_tile_guard(const _Float16* base, long laneStride,
                                       long kStride, int lane, int nvalid) {
    int r    = lane & 15;
    int ksel = (lane >> 4) << 3;
    v16h out;
#pragma unroll
    for (int i = 0; i < 16; ++i) {
        int k = ksel + (i < 8 ? i : i + 8);
        out[i] = (r < nvalid) ? base[(long)r * laneStride + (long)k * kStride]
                              : (_Float16)0.f;
    }
    return out;
}

// ---- exact top-k by ranking (matches jax.lax.top_k tie-breaking) -----------
__global__ void topk_kernel(const float* __restrict__ ar, int* __restrict__ topidx) {
    int b = blockIdx.x;
    const float* row = ar + (long)b * S_ * S_ + 1;   // attention_rollout[b,0,1:]
    for (int i = threadIdx.x; i < S_ - 1; i += blockDim.x) {
        float v = row[i];
        int rank = 0;
        for (int j = 0; j < S_ - 1; ++j) {
            float vj = row[j];
            rank += (vj > v) || (vj == v && j < i);
        }
        if (rank < NL_) topidx[b * NL_ + rank] = i + 1;
    }
}

__global__ void cvt_f32_f16(const float* __restrict__ in, _Float16* __restrict__ out, long n) {
    long i = (long)blockIdx.x * blockDim.x + threadIdx.x;
    if (i < n) out[i] = (_Float16)in[i];
}

__global__ void gather_kernel(const float* __restrict__ x, const int* __restrict__ topidx,
                              _Float16* __restrict__ xl) {
    long i = (long)blockIdx.x * blockDim.x + threadIdx.x;
    if (i >= (long)MQ_ * D_) return;
    long rowi = i / D_;
    int  d = (int)(i % D_);
    int  b = (int)(rowi / NL_), l = (int)(rowi % NL_);
    int  s = topidx[b * NL_ + l];
    xl[i] = (_Float16)x[((long)b * S_ + s) * D_ + d];
}

// ---- generic C = A (MxK, f16) * W^T (W is NxK row-major, f16) --------------
// block = 8 waves; block tile = 16(M) x 128(N); one WMMA 16x16 tile per wave.
// mode 0: outH row-major f16   mode 1: outF row-major f32
// mode 2: outH = vT[b][h][dk][s] f16 (per-head transposed V for the PV phase)
__global__ __launch_bounds__(256) void gemm_awt(const _Float16* __restrict__ A,
                                                const _Float16* __restrict__ W,
                                                _Float16* __restrict__ outH,
                                                float* __restrict__ outF,
                                                int M, int N, int K, int mode) {
    int lane = threadIdx.x & 31;
    int wave = threadIdx.x >> 5;
    int m0 = blockIdx.x * 16;
    int n0 = blockIdx.y * 128 + wave * 16;
    const _Float16* aP = A + (long)m0 * K;
    const _Float16* wP = W + (long)n0 * K;
    v8f c = {};
    for (int k0 = 0; k0 < K; k0 += 32) {
        // Branch-free prefetch of the next-next k tile (global_prefetch_b8).
        // A tail overrun of <=128B lands in the adjacent workspace buffer;
        // keeping the loop body straight-line lets the scheduler pipeline
        // load clauses across iterations (partial s_wait_loadcnt + wmma).
        __builtin_prefetch((const void*)(aP + k0 + 64), 0, 0);
        __builtin_prefetch((const void*)(wP + k0 + 64), 0, 0);
        v16h a = load_tile_k1(aP + k0, K, lane);
        v16h b = load_tile_k1(wP + k0, K, lane);
        c = __builtin_amdgcn_wmma_f32_16x16x32_f16(false, a, false, b,
                                                   (short)0, c, false, false);
    }
    int row = (lane >> 4) << 3;
    int col = lane & 15;
#pragma unroll
    for (int r = 0; r < 8; ++r) {
        int m = m0 + row + r;
        int n = n0 + col;
        if (mode == 2) {
            int b  = m >> 11, s = m & (S_ - 1);      // m = b*S + s
            int h  = n >> 6,  dk = n & (DK_ - 1);    // n = h*64 + dk
            outH[((long)(b * H_ + h) * DK_ + dk) * S_ + s] = (_Float16)c[r];
        } else if (mode == 1) {
            outF[(long)m * N + n] = c[r];
        } else {
            outH[(long)m * N + n] = (_Float16)c[r];
        }
    }
}

// ---- fused attention: QK^T/8 -> softmax -> attn out -> PV ------------------
// grid = (20 l-tiles, 12 heads, 16 batches); 8 waves; dyn LDS:
//   scH[16][2048] f16 (64KB) + part[8][16][16] f32 (8KB) = 72KB
__global__ __launch_bounds__(256) void attn_kernel(const _Float16* __restrict__ qh,
                                                   const _Float16* __restrict__ kh,
                                                   const _Float16* __restrict__ vT,
                                                   float* __restrict__ attnOut,
                                                   _Float16* __restrict__ ctxh) {
    extern __shared__ char ldsRaw[];
    _Float16* scH  = (_Float16*)ldsRaw;                        // 16*2048 f16
    float*    part = (float*)(ldsRaw + (size_t)16 * S_ * 2);   // 8*16*16 f32

    int lt = blockIdx.x, h = blockIdx.y, b = blockIdx.z;
    int lane = threadIdx.x & 31, wave = threadIdx.x >> 5;
    int l0 = lt * 16;
    int nvalid = NL_ - l0; if (nvalid > 16) nvalid = 16;

    // ---- phase 1: scores = Q K^T * (1/8) into LDS (f16)
    const _Float16* qbase = qh + (long)(b * NL_ + l0) * D_ + h * DK_;
    v16h qa0 = load_tile_guard(qbase,      D_, 1, lane, nvalid);
    v16h qa1 = load_tile_guard(qbase + 32, D_, 1, lane, nvalid);
    const _Float16* kbase = kh + (long)b * S_ * D_ + h * DK_;
    int crow = (lane >> 4) << 3, ccol = lane & 15;
    long laneRowOff = (long)(lane & 15) * D_ + ((lane >> 4) << 3);
    for (int st = wave; st < S_ / 16; st += 8) {
        const _Float16* kb = kbase + (long)(st * 16) * D_;
        // per-lane prefetch of this wave's next K tile (128 rows ahead)
        __builtin_prefetch((const void*)(kb + (long)(8 * 16) * D_ + laneRowOff), 0, 0);
        v8f c = {};
        v16h b0 = load_tile_k1(kb,      D_, lane);
        c = __builtin_amdgcn_wmma_f32_16x16x32_f16(false, qa0, false, b0, (short)0, c, false, false);
        v16h b1 = load_tile_k1(kb + 32, D_, lane);
        c = __builtin_amdgcn_wmma_f32_16x16x32_f16(false, qa1, false, b1, (short)0, c, false, false);
#pragma unroll
        for (int r = 0; r < 8; ++r)
            scH[(crow + r) * S_ + st * 16 + ccol] = (_Float16)(c[r] * 0.125f);
    }
    __syncthreads();

    // ---- phase 2: row softmax (16 lanes per row, same wave), write attention
    int rrow = threadIdx.x >> 4;
    int j    = threadIdx.x & 15;
    _Float16* srow = scH + rrow * S_;
    float mx = -3.0e38f;
    for (int cx = j; cx < S_; cx += 16) mx = fmaxf(mx, (float)srow[cx]);
#pragma unroll
    for (int o = 8; o >= 1; o >>= 1) mx = fmaxf(mx, __shfl_xor(mx, o, 16));
    float sum = 0.f;
    for (int cx = j; cx < S_; cx += 16) {
        float e = __expf((float)srow[cx] - mx);
        srow[cx] = (_Float16)e;
        sum += e;
    }
#pragma unroll
    for (int o = 8; o >= 1; o >>= 1) sum += __shfl_xor(sum, o, 16);
    float inv = 1.f / sum;
    float* aout = attnOut + ((long)(b * H_ + h) * NL_ + (l0 + rrow)) * S_;
    for (int cx = j; cx < S_; cx += 16) {
        float p = (float)srow[cx] * inv;
        srow[cx] = (_Float16)p;
        if (rrow < nvalid) aout[cx] = p;
    }
    __syncthreads();

    // ---- phase 3: context = P (LDS f16) @ V (per-head transposed, K-contig)
    int ntile = wave & 3, khalf = wave >> 2;
    const _Float16* vbase = vT + ((long)(b * H_ + h) * DK_ + ntile * 16) * S_;
    v8f c = {};
    for (int kt = 0; kt < 32; ++kt) {
        int k0 = khalf * 1024 + kt * 32;
        v16h a  = load_tile_k1(scH + k0, S_, lane);      // ds_load_b128 x2
        v16h bt = load_tile_k1(vbase + k0, S_, lane);    // global_load_b128 x2
        c = __builtin_amdgcn_wmma_f32_16x16x32_f16(false, a, false, bt, (short)0, c, false, false);
    }
#pragma unroll
    for (int r = 0; r < 8; ++r)
        part[(wave * 16 + crow + r) * 16 + ccol] = c[r];
    __syncthreads();
    if (wave < 4) {
#pragma unroll
        for (int r = 0; r < 8; ++r) {
            float s2 = part[(wave * 16 + crow + r) * 16 + ccol]
                     + part[((wave + 4) * 16 + crow + r) * 16 + ccol];
            int l = l0 + crow + r;
            if (l < NL_)
                ctxh[(long)(b * NL_ + l) * D_ + h * DK_ + ntile * 16 + ccol] = (_Float16)s2;
        }
    }
}

extern "C" void kernel_launch(void* const* d_in, const int* in_sizes, int n_in,
                              void* d_out, int out_size, void* d_ws, size_t ws_size,
                              hipStream_t stream) {
    (void)in_sizes; (void)n_in; (void)out_size; (void)ws_size;
    const float* x  = (const float*)d_in[0];
    const float* ar = (const float*)d_in[1];
    const float* Wq = (const float*)d_in[2];
    const float* Wk = (const float*)d_in[3];
    const float* Wv = (const float*)d_in[4];
    const float* Wo = (const float*)d_in[5];
    float* outp    = (float*)d_out;                       // (B, NL, D) f32
    float* attnOut = outp + (long)MQ_ * D_;               // (B, H, NL, S) f32

    char* ws = (char*)d_ws;
    size_t off = 0;
    auto alloc = [&](size_t bytes) -> void* {
        void* p = ws + off;
        off = (off + bytes + 255) & ~(size_t)255;
        return p;
    };
    int*       topidx = (int*)      alloc((size_t)B_ * NL_ * 4);
    _Float16*  xh     = (_Float16*) alloc((size_t)B_ * S_ * D_ * 2);
    _Float16*  wqh    = (_Float16*) alloc((size_t)D_ * D_ * 2);
    _Float16*  wkh    = (_Float16*) alloc((size_t)D_ * D_ * 2);
    _Float16*  wvh    = (_Float16*) alloc((size_t)D_ * D_ * 2);
    _Float16*  woh    = (_Float16*) alloc((size_t)D_ * D_ * 2);
    _Float16*  xlh    = (_Float16*) alloc((size_t)MQ_ * D_ * 2);
    _Float16*  qh     = (_Float16*) alloc((size_t)MQ_ * D_ * 2);
    _Float16*  kh     = (_Float16*) alloc((size_t)B_ * S_ * D_ * 2);
    _Float16*  vT     = (_Float16*) alloc((size_t)B_ * S_ * D_ * 2);  // (B,H,DK,S)
    _Float16*  ctxh   = (_Float16*) alloc((size_t)MQ_ * D_ * 2);

    topk_kernel<<<B_, 256, 0, stream>>>(ar, topidx);

    long nx = (long)B_ * S_ * D_;
    cvt_f32_f16<<<(unsigned)((nx + 255) / 256), 256, 0, stream>>>(x, xh, nx);
    long nw = (long)D_ * D_;
    cvt_f32_f16<<<(unsigned)((nw + 255) / 256), 256, 0, stream>>>(Wq, wqh, nw);
    cvt_f32_f16<<<(unsigned)((nw + 255) / 256), 256, 0, stream>>>(Wk, wkh, nw);
    cvt_f32_f16<<<(unsigned)((nw + 255) / 256), 256, 0, stream>>>(Wv, wvh, nw);
    cvt_f32_f16<<<(unsigned)((nw + 255) / 256), 256, 0, stream>>>(Wo, woh, nw);

    long ng = (long)MQ_ * D_;
    gather_kernel<<<(unsigned)((ng + 255) / 256), 256, 0, stream>>>(x, topidx, xlh);

    // K, V projections: (32768 x 768) @ (768 x 768)^T
    dim3 gKV((B_ * S_) / 16, D_ / 128);
    gemm_awt<<<gKV, 256, 0, stream>>>(xh, wkh, kh, nullptr, B_ * S_, D_, D_, 0);
    gemm_awt<<<gKV, 256, 0, stream>>>(xh, wvh, vT, nullptr, B_ * S_, D_, D_, 2);
    // Q projection: (4912 x 768) @ (768 x 768)^T
    dim3 gQ(MQ_ / 16, D_ / 128);
    gemm_awt<<<gQ, 256, 0, stream>>>(xlh, wqh, qh, nullptr, MQ_, D_, D_, 0);

    // fused attention
    dim3 gA(LT_, H_, B_);
    size_t smem = (size_t)16 * S_ * 2 + (size_t)8 * 16 * 16 * 4;  // 72 KB
    attn_kernel<<<gA, 256, smem, stream>>>(qh, kh, vT, attnOut, ctxh);

    // output projection: (4912 x 768) @ (768 x 768)^T -> f32 d_out
    gemm_awt<<<gQ, 256, 0, stream>>>(ctxh, woh, nullptr, outp, MQ_, D_, D_, 1);
}
